// SingleStreamMultiAttention_27882927686090
// MI455X (gfx1250) — matrix operand order repaired and checked by reading, not compiled, not run on hardware
//
#include <hip/hip_runtime.h>
#include <hip/hip_bf16.h>

// ---------------------------------------------------------------------------
// SingleStreamMultiAttention for MI455X (gfx1250, wave32, WMMA 16x16x32 f16)
//
// Pipeline:
//   1. minmax_kernel     : min/max of x_ref_attn_map rows          (tiny)
//   2. ropetab_kernel    : per-token speaker RoPE cos/sin [L][64]  (tiny)
//   3. gemm_f16split<0>  : q = x@wq^T+bq, fused RMS+RoPE -> f16 q  (155 GF)
//   4. gemm_f16split<1>  : kv = e@wkv^T+bkv, k RMS+RoPE, v^T       (6 GF)
//   5. attn_kernel       : softmax(q k^T/sqrt(hd)) v  -> fp32      (13 GF)
//   6. gemm_f16split<2>  : out = a@w_proj^T + b_proj  -> d_out     (155 GF)
//
// Projections use f16 hi/lo split (3 WMMAs / tile-pair) for ~fp32 accuracy.
// GEMM main loop uses register-prefetch pipelining: next K-step's global
// loads are issued before the current step's WMMA burst.
// ---------------------------------------------------------------------------

typedef _Float16 h8   __attribute__((ext_vector_type(8)));
typedef _Float16 h16  __attribute__((ext_vector_type(16)));
typedef float    v8f  __attribute__((ext_vector_type(8)));

constexpr int H_   = 12;
constexpr int HD   = 128;
constexpr int C_   = 1536;
constexpr int NT   = 21;
constexpr int S_   = 1560;
constexpr int B_   = 21;          // = NT (B0 == 1)
constexpr int NA   = 64;
constexpr int L_   = NT * S_;     // 32760
constexpr int LPAD = 33024;       // padded q rows (covers attn tile overhang)
constexpr int LTAB = 32768;       // padded rope-table rows (256 M-tiles * 128)

__device__ __forceinline__ h16 cat8(h8 a, h8 b) {
  return __builtin_shufflevector(a, b, 0,1,2,3,4,5,6,7,8,9,10,11,12,13,14,15);
}
__device__ __forceinline__ v8f wmma_f16(h16 a, h16 b, v8f c) {
  // v_wmma_f32_16x16x32_f16  (neg_a, A, neg_b, B, c_mod, C, reuse_a, reuse_b)
  return __builtin_amdgcn_wmma_f32_16x16x32_f16(false, a, false, b, (short)0, c,
                                                false, false);
}

// ---------------------------------------------------------------------------
// 1) min/max of the two rows of x_ref_attn_map -> red[4] = {min0,max0,min1,max1}
// ---------------------------------------------------------------------------
__global__ __launch_bounds__(256) void minmax_kernel(
    const float* __restrict__ map, float* __restrict__ red) {
  __shared__ float sn0[256], sx0[256], sn1[256], sx1[256];
  const int t = threadIdx.x;
  float mn0 = 3.4e38f, mx0 = -3.4e38f, mn1 = 3.4e38f, mx1 = -3.4e38f;
  for (int i = t; i < L_; i += 256) {
    float a = map[i], b = map[L_ + i];
    mn0 = fminf(mn0, a); mx0 = fmaxf(mx0, a);
    mn1 = fminf(mn1, b); mx1 = fmaxf(mx1, b);
  }
  sn0[t] = mn0; sx0[t] = mx0; sn1[t] = mn1; sx1[t] = mx1;
  __syncthreads();
  for (int s = 128; s > 0; s >>= 1) {
    if (t < s) {
      sn0[t] = fminf(sn0[t], sn0[t + s]); sx0[t] = fmaxf(sx0[t], sx0[t + s]);
      sn1[t] = fminf(sn1[t], sn1[t + s]); sx1[t] = fmaxf(sx1[t], sx1[t + s]);
    }
    __syncthreads();
  }
  if (t == 0) { red[0] = sn0[0]; red[1] = sx0[0]; red[2] = sn1[0]; red[3] = sx1[0]; }
}

// ---------------------------------------------------------------------------
// 2) RoPE cos/sin table for queries: cosT/sinT[l*64 + i], i = freq index
//    pos[l] = argmax row -> normalize_and_scale into (0,4) or (20,24)
// ---------------------------------------------------------------------------
__global__ __launch_bounds__(256) void ropetab_kernel(
    const float* __restrict__ map, const float* __restrict__ red,
    float* __restrict__ cosT, float* __restrict__ sinT) {
  const int idx = blockIdx.x * 256 + threadIdx.x;
  if (idx >= L_ * 64) return;
  const int l = idx >> 6, i = idx & 63;
  const float a0 = map[l], a1 = map[L_ + l];
  float pos;
  if (a0 >= a1)  // argmax picks first row on ties
    pos = (a0 - red[0]) / (red[1] - red[0] + 1e-8f) * 4.0f;
  else
    pos = (a1 - red[2]) / (red[3] - red[2] + 1e-8f) * 4.0f + 20.0f;
  const float invf = powf(10000.0f, -(float)i * (1.0f / 64.0f));
  const float f = pos * invf;
  cosT[idx] = cosf(f);
  sinT[idx] = sinf(f);
}

// ---------------------------------------------------------------------------
// 3/4/6) GEMM  C[m,n] = sum_k A[m,k] * W[n,k] + bias[n], f16 hi/lo split.
// BM=BN=128, BK=32, 256 threads = 8 waves; wave w owns rows [16w,16w+16) x
// all 128 cols (8 C fragments) so per-row reductions stay inside the wave.
// EPI: 0 = Q proj (RMS + RoPE -> f16 q), 1 = KV proj (k RMS+RoPE, v^T),
//      2 = out proj (fp32 store).
// ---------------------------------------------------------------------------
template <int EPI>
__global__ __launch_bounds__(256) void gemm_f16split(
    const float* __restrict__ A, const float* __restrict__ W,
    const float* __restrict__ bias, const int K, const int Mvalid,
    const float* __restrict__ normw,
    const float* __restrict__ cosT, const float* __restrict__ sinT,
    _Float16* __restrict__ qout, _Float16* __restrict__ kout,
    _Float16* __restrict__ vTout, float* __restrict__ fout) {
  constexpr int BM = 128, BN = 128, BK = 32, PAD = 48;
  __shared__ _Float16 sAhi[BM][PAD], sAlo[BM][PAD];
  __shared__ _Float16 sBhi[BN][PAD], sBlo[BN][PAD];

  const int tid = threadIdx.x, lane = tid & 31, wave = tid >> 5;
  const int nlane = lane & 15, halfsel = lane >> 4;
  const int mtile = blockIdx.x, ntile = blockIdx.y;

  v8f acc[8];
#pragma unroll
  for (int i = 0; i < 8; ++i)
#pragma unroll
    for (int r = 0; r < 8; ++r) acc[i][r] = 0.0f;

  // global->register staging: thread t owns 16 contiguous fp32 of one row
  const int  ldrow  = tid >> 1;
  const int  ldcol  = (tid & 1) << 4;
  const long arow   = (long)mtile * BM + ldrow;
  const bool avalid = arow < (long)Mvalid;
  const float* aptr = A + (avalid ? arow : 0) * (long)K + ldcol;
  const float* bptr = W + ((long)ntile * BN + ldrow) * (long)K + ldcol;

  float areg[16], breg[16];
#pragma unroll
  for (int j = 0; j < 16; ++j) {       // prologue: prefetch K-step 0
    areg[j] = avalid ? aptr[j] : 0.0f;
    breg[j] = bptr[j];
  }

  for (int k0 = 0; k0 < K; k0 += BK) {
    // split fp32 -> f16 hi/lo into LDS
#pragma unroll
    for (int j = 0; j < 16; ++j) {
      _Float16 ha = (_Float16)areg[j];
      sAhi[ldrow][ldcol + j] = ha;
      sAlo[ldrow][ldcol + j] = (_Float16)(areg[j] - (float)ha);
      _Float16 hb = (_Float16)breg[j];
      sBhi[ldrow][ldcol + j] = hb;
      sBlo[ldrow][ldcol + j] = (_Float16)(breg[j] - (float)hb);
    }
    __syncthreads();

    // issue next K-step's global loads NOW: they fly during the WMMA burst
    if (k0 + BK < K) {
      const int kn = k0 + BK;
#pragma unroll
      for (int j = 0; j < 16; ++j) {
        areg[j] = avalid ? aptr[kn + j] : 0.0f;
        breg[j] = bptr[kn + j];
      }
    }

    // A fragment (16x32): lane m = nlane; halves k = 16*halfsel+{0..15}
    const int mloc = wave * 16 + nlane;
    const int ka   = halfsel << 4;
    h16 ahi = cat8(*(const h8*)&sAhi[mloc][ka], *(const h8*)&sAhi[mloc][ka + 8]);
    h16 alo = cat8(*(const h8*)&sAlo[mloc][ka], *(const h8*)&sAlo[mloc][ka + 8]);
    // B fragment (32x16): lane col n = nlane; halves k = 16*halfsel+{0..15}
    const int kb = halfsel << 4;
#pragma unroll
    for (int nt = 0; nt < 8; ++nt) {
      const int nloc = nt * 16 + nlane;
      h16 bhi = *(const h16*)&sBhi[nloc][kb];
      h16 blo = *(const h16*)&sBlo[nloc][kb];
      acc[nt] = wmma_f16(ahi, bhi, acc[nt]);   // hi*hi
      acc[nt] = wmma_f16(ahi, blo, acc[nt]);   // hi*lo
      acc[nt] = wmma_f16(alo, bhi, acc[nt]);   // lo*hi
    }
    __syncthreads();
  }

  // ---- epilogue -----------------------------------------------------------
  const int nbase = ntile * BN;
#pragma unroll
  for (int nt = 0; nt < 8; ++nt) {
    float bc = bias[nbase + nt * 16 + nlane];
#pragma unroll
    for (int r = 0; r < 8; ++r) acc[nt][r] += bc;
  }

  if constexpr (EPI == 0) {
    // q: RMS over hd=128 (one N tile == one head), then speaker RoPE
    const int hh = ntile;
    float wcol[8];
#pragma unroll
    for (int nt = 0; nt < 8; ++nt) wcol[nt] = normw[nt * 16 + nlane];
#pragma unroll
    for (int r = 0; r < 8; ++r) {
      float ss = 0.0f;
#pragma unroll
      for (int nt = 0; nt < 8; ++nt) { float v = acc[nt][r]; ss += v * v; }
      ss += __shfl_xor(ss, 1); ss += __shfl_xor(ss, 2);
      ss += __shfl_xor(ss, 4); ss += __shfl_xor(ss, 8);
      const float rms = rsqrtf(ss * (1.0f / 128.0f) + 1e-6f);
      const long mrow = (long)mtile * BM + wave * 16 + r + (halfsel << 3);
      const float* cr = cosT + mrow * 64;
      const float* sr = sinT + mrow * 64;
      const bool valid = mrow < (long)Mvalid;
#pragma unroll
      for (int nt = 0; nt < 8; ++nt) {
        const int d = nt * 16 + nlane;
        float v = acc[nt][r] * rms * wcol[nt];
        float p = __shfl_xor(v, 1);  // rotate_half partner (adjacent lane)
        float cv = cr[d >> 1], sv = sr[d >> 1];
        float o = (d & 1) ? fmaf(v, cv, p * sv) : fmaf(v, cv, -p * sv);
        if (valid) qout[((size_t)hh * LPAD + (size_t)mrow) * HD + d] = (_Float16)o;
      }
    }
  } else if constexpr (EPI == 1) {
    const int two = ntile / H_, hh = ntile % H_;   // kv.reshape(...,2,H,hd)
    if (two == 0) {  // keys: RMS + fixed-position RoPE (2.0 / 22.0)
      float wcol[8];
#pragma unroll
      for (int nt = 0; nt < 8; ++nt) wcol[nt] = normw[nt * 16 + nlane];
#pragma unroll
      for (int r = 0; r < 8; ++r) {
        float ss = 0.0f;
#pragma unroll
        for (int nt = 0; nt < 8; ++nt) { float v = acc[nt][r]; ss += v * v; }
        ss += __shfl_xor(ss, 1); ss += __shfl_xor(ss, 2);
        ss += __shfl_xor(ss, 4); ss += __shfl_xor(ss, 8);
        const float rms = rsqrtf(ss * (1.0f / 128.0f) + 1e-6f);
        const long m = (long)mtile * BM + wave * 16 + r + (halfsel << 3);
        const bool valid = m < (long)Mvalid;
        const int bidx = (int)(m >> 6), na = (int)(m & 63);
        const float pos = (na < 32) ? 2.0f : 22.0f;
#pragma unroll
        for (int nt = 0; nt < 8; ++nt) {
          const int d = nt * 16 + nlane;
          float v = acc[nt][r] * rms * wcol[nt];
          float p = __shfl_xor(v, 1);
          float fr = pos * powf(10000.0f, -(float)(d >> 1) * (1.0f / 64.0f));
          float cv = cosf(fr), sv = sinf(fr);
          float o = (d & 1) ? fmaf(v, cv, p * sv) : fmaf(v, cv, -p * sv);
          if (valid)
            kout[(((size_t)hh * B_ + bidx) * NA + na) * HD + d] = (_Float16)o;
        }
      }
    } else {  // values: store transposed v^T[h][b][d][na]
#pragma unroll
      for (int r = 0; r < 8; ++r) {
        const long m = (long)mtile * BM + wave * 16 + r + (halfsel << 3);
        const bool valid = m < (long)Mvalid;
        const int bidx = (int)(m >> 6), na = (int)(m & 63);
#pragma unroll
        for (int nt = 0; nt < 8; ++nt) {
          const int d = nt * 16 + nlane;
          if (valid)
            vTout[(((size_t)hh * B_ + bidx) * HD + d) * NA + na] =
                (_Float16)acc[nt][r];
        }
      }
    }
  } else {  // EPI == 2: final projection, fp32 store
#pragma unroll
    for (int r = 0; r < 8; ++r) {
      const long m = (long)mtile * BM + wave * 16 + r + (halfsel << 3);
      if (m < (long)Mvalid) {
#pragma unroll
        for (int nt = 0; nt < 8; ++nt)
          fout[(size_t)m * C_ + nbase + nt * 16 + nlane] = acc[nt][r];
      }
    }
  }
}

// ---------------------------------------------------------------------------
// 5) attention: per wave 16 query rows x (Na=64 keys), softmax, @ v
// grid = (ceil(S/128), H, B), 256 threads = 8 waves
// ---------------------------------------------------------------------------
__global__ __launch_bounds__(256) void attn_kernel(
    const _Float16* __restrict__ qbuf, const _Float16* __restrict__ kbuf,
    const _Float16* __restrict__ vT, float* __restrict__ aout) {
  __shared__ _Float16 sp[8][16][72];  // per-wave 16x64 attn tile (+pad)
  const int tid = threadIdx.x, lane = tid & 31, wave = tid >> 5;
  const int nlane = lane & 15, halfsel = lane >> 4;
  const int hh = blockIdx.y, b = blockIdx.z;
  const int s0 = blockIdx.x * 128 + wave * 16;
  const long l0 = (long)b * S_ + s0;

  // q A fragments over K = hd = 128 (4 chunks of 32)
  h16 aq[4];
  const _Float16* qr = qbuf + ((size_t)hh * LPAD + l0 + nlane) * HD;
#pragma unroll
  for (int kb = 0; kb < 4; ++kb) {
    const int ko = kb * 32 + (halfsel << 4);
    aq[kb] = cat8(*(const h8*)(qr + ko), *(const h8*)(qr + ko + 8));
  }

  // scores = q @ k^T  (4 N tiles of 16 keys)
  v8f sc[4];
#pragma unroll
  for (int nt = 0; nt < 4; ++nt) {
    v8f c;
#pragma unroll
    for (int r = 0; r < 8; ++r) c[r] = 0.0f;
    const _Float16* kr =
        kbuf + (((size_t)hh * B_ + b) * NA + nt * 16 + nlane) * HD;
#pragma unroll
    for (int kb = 0; kb < 4; ++kb) {
      h16 bk = *(const h16*)(kr + kb * 32 + (halfsel << 4));
      c = wmma_f16(aq[kb], bk, c);
    }
    sc[nt] = c;
  }

  // row-wise softmax over 64 keys (cols live in lane halves + 4 fragments)
  const float scale = 0.08838834764831845f;  // hd^-0.5
#pragma unroll
  for (int r = 0; r < 8; ++r) {
    float v0 = sc[0][r] * scale, v1 = sc[1][r] * scale;
    float v2 = sc[2][r] * scale, v3 = sc[3][r] * scale;
    float mx = fmaxf(fmaxf(v0, v1), fmaxf(v2, v3));
    mx = fmaxf(mx, __shfl_xor(mx, 1)); mx = fmaxf(mx, __shfl_xor(mx, 2));
    mx = fmaxf(mx, __shfl_xor(mx, 4)); mx = fmaxf(mx, __shfl_xor(mx, 8));
    float e0 = expf(v0 - mx), e1 = expf(v1 - mx);
    float e2 = expf(v2 - mx), e3 = expf(v3 - mx);
    float sum = e0 + e1 + e2 + e3;
    sum += __shfl_xor(sum, 1); sum += __shfl_xor(sum, 2);
    sum += __shfl_xor(sum, 4); sum += __shfl_xor(sum, 8);
    const float inv = 1.0f / sum;
    const int row = r + (halfsel << 3);
    sp[wave][row][0 * 16 + nlane] = (_Float16)(e0 * inv);
    sp[wave][row][1 * 16 + nlane] = (_Float16)(e1 * inv);
    sp[wave][row][2 * 16 + nlane] = (_Float16)(e2 * inv);
    sp[wave][row][3 * 16 + nlane] = (_Float16)(e3 * inv);
  }
  __syncthreads();  // uniform for all 8 waves (no early exits)

  // re-load attn weights in A-fragment layout (K = 64 -> 2 chunks)
  h16 ap[2];
  const _Float16* ar = &sp[wave][nlane][0];
#pragma unroll
  for (int kb2 = 0; kb2 < 2; ++kb2) {
    const int no = kb2 * 32 + (halfsel << 4);
    ap[kb2] = cat8(*(const h8*)(ar + no), *(const h8*)(ar + no + 8));
  }

  // out = attn @ v  (8 N tiles over hd = 128), store fp32 [b*S+s][h*128+d]
#pragma unroll
  for (int nt2 = 0; nt2 < 8; ++nt2) {
    v8f c;
#pragma unroll
    for (int r = 0; r < 8; ++r) c[r] = 0.0f;
    const _Float16* vr =
        vT + (((size_t)hh * B_ + b) * HD + nt2 * 16 + nlane) * NA;
#pragma unroll
    for (int kb2 = 0; kb2 < 2; ++kb2) {
      h16 bv = *(const h16*)(vr + kb2 * 32 + (halfsel << 4));
      c = wmma_f16(ap[kb2], bv, c);
    }
#pragma unroll
    for (int r = 0; r < 8; ++r) {
      const int srow = s0 + r + (halfsel << 3);
      if (srow < S_)
        aout[((size_t)b * S_ + srow) * C_ + hh * HD + nt2 * 16 + nlane] = c[r];
    }
  }
}

// ---------------------------------------------------------------------------
extern "C" void kernel_launch(void* const* d_in, const int* in_sizes, int n_in,
                              void* d_out, int out_size, void* d_ws,
                              size_t ws_size, hipStream_t stream) {
  const float* x     = (const float*)d_in[0];
  const float* enc   = (const float*)d_in[1];   // [21,64,768] rows (b,na)
  const float* map   = (const float*)d_in[2];   // [2, L]
  const float* wq    = (const float*)d_in[3];
  const float* bq    = (const float*)d_in[4];
  const float* wkv   = (const float*)d_in[5];
  const float* bkv   = (const float*)d_in[6];
  const float* wproj = (const float*)d_in[7];
  const float* bproj = (const float*)d_in[8];
  const float* qnw   = (const float*)d_in[9];
  const float* knw   = (const float*)d_in[10];
  float* out = (float*)d_out;

  // workspace carve-up (~330 MB total)
  char* ws = (char*)d_ws;
  size_t off = 0;
  auto carve = [&](size_t bytes) -> void* {
    void* p = ws + off;
    off = (off + bytes + 255) & ~(size_t)255;
    return p;
  };
  float*     red  = (float*)carve(4 * sizeof(float));
  float*     cosT = (float*)carve((size_t)LTAB * 64 * sizeof(float));
  float*     sinT = (float*)carve((size_t)LTAB * 64 * sizeof(float));
  _Float16*  qb   = (_Float16*)carve((size_t)H_ * LPAD * HD * sizeof(_Float16));
  _Float16*  kb   = (_Float16*)carve((size_t)H_ * B_ * NA * HD * sizeof(_Float16));
  _Float16*  vb   = (_Float16*)carve((size_t)H_ * B_ * HD * NA * sizeof(_Float16));
  float*     ao   = (float*)carve((size_t)LTAB * C_ * sizeof(float));

  // 1) row min/max for normalize_and_scale
  minmax_kernel<<<1, 256, 0, stream>>>(map, red);
  // 2) per-token RoPE cos/sin table
  ropetab_kernel<<<(L_ * 64 + 255) / 256, 256, 0, stream>>>(map, red, cosT, sinT);
  // 3) q projection + RMS + RoPE  (M = 32760, K = 1536, N = 1536)
  gemm_f16split<0><<<dim3(256, 12), 256, 0, stream>>>(
      x, wq, bq, C_, L_, qnw, cosT, sinT, qb, nullptr, nullptr, nullptr);
  // 4) kv projection + k RMS/RoPE + v^T  (M = 1344, K = 768, N = 3072)
  gemm_f16split<1><<<dim3(11, 24), 256, 0, stream>>>(
      enc, wkv, bkv, 768, B_ * NA, knw, nullptr, nullptr, nullptr, kb, vb,
      nullptr);
  // 5) cross attention
  attn_kernel<<<dim3(13, H_, B_), 256, 0, stream>>>(qb, kb, vb, ao);
  // 6) output projection  (M = 32760, K = 1536, N = 1536) -> d_out
  gemm_f16split<2><<<dim3(256, 12), 256, 0, stream>>>(
      ao, wproj, bproj, C_, L_, nullptr, nullptr, nullptr, nullptr, nullptr,
      nullptr, out);
}
